// HIGnnInterface_47837345742946
// MI455X (gfx1250) — matrix operand order repaired and checked by reading, not compiled, not run on hardware
//
#include <hip/hip_runtime.h>
#include <hip/hip_bf16.h>
#include <math.h>

typedef __attribute__((ext_vector_type(16))) __bf16 v16bf;
typedef __attribute__((ext_vector_type(8)))  float  v8f;

#define CCH 128
#define EPSV 1e-4f

// ---------------- zero workspace ----------------
__global__ void zero_f32(float* __restrict__ p, long long n) {
  long long i = (long long)blockIdx.x * blockDim.x + threadIdx.x;
  long long stride = (long long)gridDim.x * blockDim.x;
  for (; i < n; i += stride) p[i] = 0.0f;
}

// ---------------- degree: atomic count into deg[col] ----------------
__global__ void degree_kernel(const long long* __restrict__ coli,
                              float* __restrict__ deg, int E) {
  int e = blockIdx.x * blockDim.x + threadIdx.x;
  if (e < E) {
    __builtin_prefetch(coli + e + 4096, 0, 0);   // global_prefetch_b8
    unsafeAtomicAdd(&deg[coli[e]], 1.0f);        // global_atomic_add_f32
  }
}

// ---------------- deg -> deg^-1/2 in place (0 stays 0) ----------------
__global__ void deginv_kernel(float* __restrict__ deg, int N) {
  int n = blockIdx.x * blockDim.x + threadIdx.x;
  if (n < N) {
    float d = deg[n];
    deg[n] = (d > 0.0f) ? rsqrtf(d) : 0.0f;
  }
}

// ---------------- scatter: agg[col] += x[row], one wave per edge ----------------
__global__ __launch_bounds__(256)
void scatter_kernel(const float* __restrict__ x,
                    const long long* __restrict__ rowi,
                    const long long* __restrict__ coli,
                    float* __restrict__ agg, int E) {
  int lane = threadIdx.x & 31;
  int e = blockIdx.x * (blockDim.x >> 5) + (threadIdx.x >> 5);
  if (e >= E) return;
  __builtin_prefetch(rowi + e + 2048, 0, 0);
  __builtin_prefetch(coli + e + 2048, 0, 0);
  long long r = rowi[e];
  long long c = coli[e];
  const float4 v = *(const float4*)(x + r * CCH + lane * 4);  // L2-resident gather
  float* dst = agg + c * CCH + lane * 4;
  unsafeAtomicAdd(dst + 0, v.x);
  unsafeAtomicAdd(dst + 1, v.y);
  unsafeAtomicAdd(dst + 2, v.z);
  unsafeAtomicAdd(dst + 3, v.w);
}

// ---------------- MP weight normalization -> bf16, one wave per output row ----------------
__global__ void wprep_kernel(const float* __restrict__ w, __bf16* __restrict__ wbf) {
  int wid  = (blockIdx.x * blockDim.x + threadIdx.x) >> 5;   // row 0..127
  int lane = threadIdx.x & 31;
  if (wid >= CCH) return;
  const float* src = w + wid * CCH;
  float4 v = *(const float4*)(src + lane * 4);
  float ss = v.x * v.x + v.y * v.y + v.z * v.z + v.w * v.w;
  #pragma unroll
  for (int o = 16; o >= 1; o >>= 1) ss += __shfl_xor(ss, o, 32);
  const float rsf = 0.08838834764831845f;        // 1/sqrt(128)
  float scale = rsf / (EPSV + sqrtf(ss) * rsf);  // (1/sqrt(fan_in)) / (eps + ||w||/sqrt(fan_in))
  __bf16* d = wbf + wid * CCH + lane * 4;
  d[0] = (__bf16)(v.x * scale);
  d[1] = (__bf16)(v.y * scale);
  d[2] = (__bf16)(v.z * scale);
  d[3] = (__bf16)(v.w * scale);
}

// ---------------- async global -> LDS copy (gfx1250, ASYNCcnt path) ----------------
__device__ __forceinline__ void async_copy_b128(unsigned int lds_off, const void* gptr) {
  // VDST = LDS byte address VGPR, VADDR = 64-bit global address, GV mode.
  asm volatile("global_load_async_to_lds_b128 %0, %1, off"
               :: "v"(lds_off), "v"(gptr)
               : "memory");
}

__device__ __forceinline__ void wait_asynccnt0() {
  asm volatile("s_wait_asynccnt 0x0" ::: "memory");
}

// ---------------- fused GEMM (bf16 WMMA) + mp_sum + deg select ----------------
// One wave per 16-row strip. Weights staged in LDS via async loads; B fragments
// come from ds_load, A fragments (x, deg_inv*agg) packed from global f32.
__global__ __launch_bounds__(256)
void gemm_combine_kernel(const float* __restrict__ x,
                         const float* __restrict__ agg,
                         const float* __restrict__ dinv,   // deg^-1/2, 0 if deg==0
                         const __bf16* __restrict__ wboth, // [wnei | wres] bf16, contiguous 64KB
                         float* __restrict__ out, int N) {
  // 64 KB of LDS: wlds[0..16383] = wnei', wlds[16384..32767] = wres'
  __shared__ __bf16 wlds[2 * CCH * CCH];

  // Cooperative async fill (must run in every wave before any exit: barrier below).
  {
    const char* src = (const char*)wboth;
    unsigned int base = (unsigned int)(unsigned long long)(&wlds[0]);
    #pragma unroll
    for (int i = 0; i < 16; ++i) {
      unsigned int off = (threadIdx.x + i * 256u) * 16u;   // 256 thr * 16 it * 16B = 64KB
      async_copy_b128(base + off, src + off);
    }
    wait_asynccnt0();
    __syncthreads();
  }

  const int lane  = threadIdx.x & 31;
  const int strip = blockIdx.x * (blockDim.x >> 5) + (threadIdx.x >> 5);
  const int row0  = strip * 16;
  if (row0 >= N) return;

  const __bf16* wneil = &wlds[0];          // LDS-resident normalized weights
  const __bf16* wresl = &wlds[CCH * CCH];

  const int mA   = lane & 15;          // A-matrix row (ISA 16-bit A layout: M = lane%16)
  const int half = (lane >> 4) & 1;

  // Build A fragments for x and (agg * deg_inv), all 4 K-tiles.
  v16bf xf[4], af[4];
  const float* xrow = x   + (long long)(row0 + mA) * CCH;
  const float* arow = agg + (long long)(row0 + mA) * CCH;
  const float  dv   = dinv[row0 + mA];
  #pragma unroll
  for (int kt = 0; kt < 4; ++kt) {
    const int kb = kt * 32;
    #pragma unroll
    for (int i = 0; i < 8; ++i) {
      // ISA 7.12.2: VGPR i pair covers K = (i>=4)*16 + half*8 + (i%4)*2 + {0,1}
      int k = kb + ((i >> 2) << 4) + (half << 3) + ((i & 3) << 1);
      xf[kt][2 * i]     = (__bf16)xrow[k];
      xf[kt][2 * i + 1] = (__bf16)xrow[k + 1];
      af[kt][2 * i]     = (__bf16)(arow[k] * dv);
      af[kt][2 * i + 1] = (__bf16)(arow[k + 1] * dv);
    }
  }

  // deg>0 flags for this lane's 8 output rows (C/D layout: m = r + half*8)
  bool dpos[8];
  #pragma unroll
  for (int r = 0; r < 8; ++r) dpos[r] = dinv[row0 + r + half * 8] > 0.0f;

  const float inv_sqrt2 = 0.70710678118654752f;
  const int   n16 = lane & 15;         // B/D column (N = lane%16)

  #pragma unroll
  for (int jt = 0; jt < 8; ++jt) {
    v8f accR = {};
    v8f accN = {};
    const int j = jt * 16 + n16;
    #pragma unroll
    for (int kt = 0; kt < 4; ++kt) {
      // B fragment from LDS: VGPR v pair covers K = (v>=4)*16 + half*8 + (v%4)*2 + {0,1}; B[k][j] = W'[j][k]
      v16bf bR, bN;
      const int kb = kt * 32 + half * 8;
      #pragma unroll
      for (int vv = 0; vv < 8; ++vv) {
        int k = kb + ((vv >> 2) << 4) + 2 * (vv & 3);
        bR[2 * vv]     = wresl[j * CCH + k];
        bR[2 * vv + 1] = wresl[j * CCH + k + 1];
        bN[2 * vv]     = wneil[j * CCH + k];
        bN[2 * vv + 1] = wneil[j * CCH + k + 1];
      }
      accR = __builtin_amdgcn_wmma_f32_16x16x32_bf16(false, xf[kt], false, bR,
                                                     (short)0, accR, false, false);
      accN = __builtin_amdgcn_wmma_f32_16x16x32_bf16(false, af[kt], false, bN,
                                                     (short)0, accN, false, false);
    }
    // mp_sum + deg>0 select, store (C/D layout: VGPR r -> row r + half*8, col lane%16)
    #pragma unroll
    for (int r = 0; r < 8; ++r) {
      float res = accR[r];
      float val = dpos[r] ? (res + accN[r]) * inv_sqrt2 : res;
      out[(long long)(row0 + r + half * 8) * CCH + j] = val;
    }
  }
}

extern "C" void kernel_launch(void* const* d_in, const int* in_sizes, int n_in,
                              void* d_out, int out_size, void* d_ws, size_t ws_size,
                              hipStream_t stream) {
  const float*     x    = (const float*)d_in[0];
  const long long* ei   = (const long long*)d_in[1];   // int64 per reference
  const float*     wnei = (const float*)d_in[2];
  const float*     wres = (const float*)d_in[3];
  float*           out  = (float*)d_out;

  const int N = in_sizes[0] / CCH;
  const int E = in_sizes[1] / 2;
  const long long* rowi = ei;       // source
  const long long* coli = ei + E;   // target

  // workspace layout: [agg N*128 f32][deg N f32][wnei bf16 128*128][wres bf16 128*128]
  float*  agg   = (float*)d_ws;
  float*  deg   = agg + (long long)N * CCH;
  __bf16* wneib = (__bf16*)(deg + N);
  __bf16* wresb = wneib + CCH * CCH;

  const long long zn = (long long)N * CCH + N;
  zero_f32<<<2048, 256, 0, stream>>>(agg, zn);

  degree_kernel<<<(E + 255) / 256, 256, 0, stream>>>(coli, deg, E);
  deginv_kernel<<<(N + 255) / 256, 256, 0, stream>>>(deg, N);

  scatter_kernel<<<(E + 7) / 8, 256, 0, stream>>>(x, rowi, coli, agg, E);

  wprep_kernel<<<16, 256, 0, stream>>>(wnei, wneib);   // 128 waves
  wprep_kernel<<<16, 256, 0, stream>>>(wres, wresb);

  const int strips = (N + 15) / 16;
  gemm_combine_kernel<<<(strips + 7) / 8, 256, 0, stream>>>(x, agg, deg, wneib,
                                                            out, N);
}
